// Model_1778116460896
// MI455X (gfx1250) — matrix operand order repaired and checked by reading, not compiled, not run on hardware
//
#include <hip/hip_runtime.h>
#include <hip/hip_bf16.h>
#include <math.h>

#define N_NODES 100000
#define N_EDGES 1600000
#define F_IN    15
#define H_DIM   64
#define OUT_DIM 14
#define BN_EPS  1e-5f

typedef __attribute__((ext_vector_type(16))) _Float16 v16h;
typedef __attribute__((ext_vector_type(8)))  float    v8f;

// ---------------------------------------------------------------- utilities
__global__ void k_zero(float* __restrict__ p, int n) {
    int i = blockIdx.x * blockDim.x + threadIdx.x;
    if (i < n) p[i] = 0.f;
}

// deg[dst] += w  (segment_sum of edge weights)
__global__ void k_deg(const int* __restrict__ ei, const float* __restrict__ ew,
                      float* __restrict__ deg) {
    int e = blockIdx.x * blockDim.x + threadIdx.x;
    if (e < N_EDGES) atomicAdd(&deg[ei[N_EDGES + e]], ew[e]);
}

__global__ void k_dinv(const float* __restrict__ deg, float* __restrict__ dinv,
                       float* __restrict__ dinv2) {
    int i = blockIdx.x * blockDim.x + threadIdx.x;
    if (i < N_NODES) {
        float d = deg[i] + 1.f;
        dinv[i]  = rsqrtf(d);
        dinv2[i] = 1.f / d;
    }
}

// --------------------------------------------- weight pre-pack (f32 -> f16)
// Packs a weight matrix into the exact WMMA 16-bit B-operand lane layout:
//   Bp[((tile*chunks + kc)*32 + lane)*16 + i]  holds
//   B[k][n] with n = tile*16 + (lane&15),  k = kc*32 + (lane<16 ? 0:8) + (i&7) + ((i&8)<<1)
// so the GEMM inner loop does ONE 32-byte v16h load per tile per k-chunk.
// BT=0: W is K x Ncols (B = W);  BT=1: W is Ncols x K (B = W^T).
__global__ void k_pack_b(const float* __restrict__ W, _Float16* __restrict__ Bp,
                         int K, int ldw, int Ncols, int BT) {
    int idx = blockIdx.x * blockDim.x + threadIdx.x;
    int chunks = (K + 31) >> 5;
    int total = (Ncols >> 4) * chunks * 32 * 16;
    if (idx >= total) return;
    int i    = idx & 15;
    int lane = (idx >> 4) & 31;
    int kc   = (idx >> 9) % chunks;
    int tile = (idx >> 9) / chunks;
    int n = (tile << 4) + (lane & 15);
    int k = (kc << 5) + ((lane < 16) ? 0 : 8) + (i & 7) + ((i & 8) << 1);
    float v = 0.f;
    if (k < K) v = BT ? W[n * ldw + k] : W[k * ldw + n];
    Bp[idx] = (_Float16)v;
}

// A-tile fragment load (ISA 16-bit A layout), K multiple of 32, 16B aligned:
// four float4 loads + packed converts.
__device__ __forceinline__ v16h load_a_vec(const float* __restrict__ A,
                                           int rowBase, int kb, int aK0) {
    const float* p = A + rowBase + kb + aK0;
    float4 f0 = *(const float4*)(p);
    float4 f1 = *(const float4*)(p + 4);
    float4 f2 = *(const float4*)(p + 16);
    float4 f3 = *(const float4*)(p + 20);
    v16h a;
    a[0] = (_Float16)f0.x;  a[1] = (_Float16)f0.y;
    a[2] = (_Float16)f0.z;  a[3] = (_Float16)f0.w;
    a[4] = (_Float16)f1.x;  a[5] = (_Float16)f1.y;
    a[6] = (_Float16)f1.z;  a[7] = (_Float16)f1.w;
    a[8]  = (_Float16)f2.x; a[9]  = (_Float16)f2.y;
    a[10] = (_Float16)f2.z; a[11] = (_Float16)f2.w;
    a[12] = (_Float16)f3.x; a[13] = (_Float16)f3.y;
    a[14] = (_Float16)f3.z; a[15] = (_Float16)f3.w;
    return a;
}

// ------------------------------------------------------- WMMA GEMM (packed B)
// C[N_NODES x 64] = A[N_NODES x K] * B[K x 64].  One wave computes a 16x64
// strip (4 n-tiles) so the A fragment is reused by 4 WMMAs per k-chunk.
// KPAD=true: K not a multiple of 32 -> guarded scalar A loads (layer 1, K=15).
template <bool KPAD>
__global__ void k_gemm_wmma(const float* __restrict__ A, const _Float16* __restrict__ Bp,
                            float* __restrict__ C, int K, int lda, int ldc) {
    const int mTile = blockIdx.x * (blockDim.x >> 5) + (threadIdx.x >> 5);
    const int lane  = threadIdx.x & 31;
    if (mTile >= (N_NODES >> 4)) return;  // wave-uniform

    const int l    = lane & 15;
    const int rowB = ((mTile << 4) + l) * lda;
    const int aK0  = (lane < 16) ? 0 : 8;
    const int chunks = (K + 31) >> 5;

    v8f acc[4] = {v8f{}, v8f{}, v8f{}, v8f{}};
    for (int kc = 0; kc < chunks; ++kc) {
        const int kb = kc << 5;
        v16h a;
        if (KPAD) {
#pragma unroll
            for (int i = 0; i < 8; ++i) {
                int k0 = kb + aK0 + i;
                int k1 = kb + aK0 + 16 + i;
                a[i]     = (k0 < K) ? (_Float16)A[rowB + k0] : (_Float16)0.f;
                a[8 + i] = (k1 < K) ? (_Float16)A[rowB + k1] : (_Float16)0.f;
            }
        } else {
            a = load_a_vec(A, rowB, kb, aK0);
        }
#pragma unroll
        for (int nt = 0; nt < 4; ++nt) {
            const v16h b = *(const v16h*)(Bp + (((nt * chunks + kc) * 32 + lane) << 4));
            acc[nt] = __builtin_amdgcn_wmma_f32_16x16x32_f16(false, a, false, b,
                                                             (short)0, acc[nt], false, false);
        }
    }
#pragma unroll
    for (int nt = 0; nt < 4; ++nt) {
        const int ccol = (nt << 4) + l;
#pragma unroll
        for (int v = 0; v < 8; ++v) {
            int crow = (mTile << 4) + v + ((lane >= 16) ? 8 : 0);
            C[crow * ldc + ccol] = acc[nt][v];
        }
    }
}

// ------------------------------------------- GCN message passing (scatter)
// agg[dst,:] += (dinv[src]*w*dinv[dst]) * xw[src,:]
// 16 lanes per edge, float4 per lane -> 256B gather (L2-resident), f32 atomics.
__global__ void k_edge_scatter(const int* __restrict__ ei, const float* __restrict__ ew,
                               const float* __restrict__ dinv,
                               const float* __restrict__ xw, float* __restrict__ agg) {
    int t = blockIdx.x * blockDim.x + threadIdx.x;
    int e = t >> 4;
    if (e >= N_EDGES) return;
    int c4 = (t & 15) << 2;
    int s = ei[e];
    int d = ei[N_EDGES + e];
    float nrm = dinv[s] * ew[e] * dinv[d];
    const float4 v = *(const float4*)(xw + s * H_DIM + c4);
    float* o = agg + d * H_DIM + c4;
    atomicAdd(o + 0, nrm * v.x);
    atomicAdd(o + 1, nrm * v.y);
    atomicAdd(o + 2, nrm * v.z);
    atomicAdd(o + 3, nrm * v.w);
}

// ------------------- GCN epilogue: self loop + bias + ReLU + BN statistics
__global__ void k_post_gcn(float* __restrict__ hbuf /* agg in, h_pre out */,
                           const float* __restrict__ xw, const float* __restrict__ dinv2,
                           const float* __restrict__ bias,
                           float* __restrict__ colsum, float* __restrict__ colsq) {
    __shared__ float s_sum[H_DIM];
    __shared__ float s_sq[H_DIM];
    int tid = threadIdx.x;
    if (tid < H_DIM) { s_sum[tid] = 0.f; s_sq[tid] = 0.f; }
    __syncthreads();
    int idx = blockIdx.x * blockDim.x + tid;
    if (idx < N_NODES * H_DIM) {
        int n = idx >> 6, c = idx & 63;
        float val = hbuf[idx] + dinv2[n] * xw[idx] + bias[c];
        val = fmaxf(val, 0.f);
        hbuf[idx] = val;
        atomicAdd(&s_sum[c], val);
        atomicAdd(&s_sq[c], val * val);
    }
    __syncthreads();
    if (tid < H_DIM) {
        atomicAdd(&colsum[tid], s_sum[tid]);
        atomicAdd(&colsq[tid], s_sq[tid]);
    }
}

// BN apply, writing into the concatenated r buffer at column offset
__global__ void k_bn_apply(const float* __restrict__ hbuf,
                           const float* __restrict__ colsum, const float* __restrict__ colsq,
                           const float* __restrict__ gamma, const float* __restrict__ beta,
                           float* __restrict__ rout, int ldr, int colOff) {
    int idx = blockIdx.x * blockDim.x + threadIdx.x;
    if (idx >= N_NODES * H_DIM) return;
    int n = idx >> 6, c = idx & 63;
    const float invN = 1.f / (float)N_NODES;
    float mu  = colsum[c] * invN;
    float var = colsq[c] * invN - mu * mu;
    float v = (hbuf[idx] - mu) * rsqrtf(var + BN_EPS) * gamma[c] + beta[c];
    rout[n * ldr + colOff + c] = v;
}

// ------------------------------------------ fused LSTM step (h0 = c0 = 0)
// g = inp @ W_ih^T + b_ih + b_hh ; torch gate order i|f|chat|o; f-gate dead.
// W_ih pre-packed as B with Ncols = 4H (16 tiles): gate tile bases i=0,
// chat=8, o=12.  The packed weight buffer (<= 64 KB) is staged into LDS once
// per workgroup via CDNA5 async-to-LDS b128 copies (ASYNCcnt +
// s_wait_asynccnt), then the k-loop feeds WMMA B operands from LDS.
__global__ void k_lstm_wmma(const float* __restrict__ inp, int K, int lda,
                            const _Float16* __restrict__ Bp,
                            const float* __restrict__ b_ih, const float* __restrict__ b_hh,
                            float* __restrict__ hh) {
    __shared__ _Float16 sB[16 * 4 * 32 * 16];  // 64 KB (chunks<=4)

    const int chunks = K >> 5;  // K = 128 or 64

    // ---- async stage: whole packed weight buffer -> LDS (16B granules)
    {
        const int n16 = 16 * chunks * 32 * 2;  // total 16B granules (1024*chunks)
        unsigned sBase = (unsigned)(unsigned long long)(&sB[0]);
        for (int u = threadIdx.x; u < n16; u += blockDim.x) {
            unsigned ldsAddr = sBase + ((unsigned)u << 4);
            const char* g = (const char*)Bp + ((unsigned long long)u << 4);
            asm volatile("global_load_async_to_lds_b128 %0, %1, off"
                         :: "v"(ldsAddr), "v"(g) : "memory");
        }
        asm volatile("s_wait_asynccnt 0x0" ::: "memory");
    }
    __syncthreads();

    const int wave = blockIdx.x * (blockDim.x >> 5) + (threadIdx.x >> 5);
    const int lane = threadIdx.x & 31;
    const int mTile = wave >> 2;        // tilesN = H/16 = 4
    const int nTile = wave & 3;
    if (mTile >= (N_NODES >> 4)) return;  // wave-uniform (never taken: exact grid)

    const int l    = lane & 15;
    const int rowB = ((mTile << 4) + l) * lda;
    const int aK0  = (lane < 16) ? 0 : 8;

    const int ti = (0 + nTile) * chunks;   // gate i    (cols [0,H))
    const int tc = (8 + nTile) * chunks;   // gate chat (cols [2H,3H))
    const int to = (12 + nTile) * chunks;  // gate o    (cols [3H,4H))

    v8f ai = {}, ac = {}, ao = {};
    for (int kc = 0; kc < chunks; ++kc) {
        const v16h a  = load_a_vec(inp, rowB, kc << 5, aK0);
        const v16h bi = *(const v16h*)(sB + (((ti + kc) * 32 + lane) << 4));
        const v16h bc = *(const v16h*)(sB + (((tc + kc) * 32 + lane) << 4));
        const v16h bo = *(const v16h*)(sB + (((to + kc) * 32 + lane) << 4));
        ai = __builtin_amdgcn_wmma_f32_16x16x32_f16(false, a, false, bi, (short)0, ai, false, false);
        ac = __builtin_amdgcn_wmma_f32_16x16x32_f16(false, a, false, bc, (short)0, ac, false, false);
        ao = __builtin_amdgcn_wmma_f32_16x16x32_f16(false, a, false, bo, (short)0, ao, false, false);
    }

    const int ccol = (nTile << 4) + l;
    const float bii = b_ih[ccol]             + b_hh[ccol];
    const float bcc = b_ih[2 * H_DIM + ccol] + b_hh[2 * H_DIM + ccol];
    const float boo = b_ih[3 * H_DIM + ccol] + b_hh[3 * H_DIM + ccol];
#pragma unroll
    for (int v = 0; v < 8; ++v) {
        int crow = (mTile << 4) + v + ((lane >= 16) ? 8 : 0);
        float ig = 1.f / (1.f + __expf(-(ai[v] + bii)));
        float cg = tanhf(ac[v] + bcc);
        float og = 1.f / (1.f + __expf(-(ao[v] + boo)));
        hh[crow * H_DIM + ccol] = og * tanhf(ig * cg);
    }
}

// ------------------------------------ output projection: relu(feats) @ Wl + bl
__global__ void k_final(const float* __restrict__ hh1, const float* __restrict__ hh2,
                        const float* __restrict__ x, const float* __restrict__ Wl,
                        const float* __restrict__ bl, float* __restrict__ out) {
    int idx = blockIdx.x * blockDim.x + threadIdx.x;
    if (idx >= N_NODES * OUT_DIM) return;
    int n = idx / OUT_DIM, o = idx % OUT_DIM;
    const float* h1 = hh1 + n * H_DIM;
    const float* h2 = hh2 + n * H_DIM;
    const float* xr = x + n * F_IN;
    float acc = bl[o];
#pragma unroll 8
    for (int k = 0; k < H_DIM; ++k) acc += fmaxf(h1[k], 0.f) * Wl[k * OUT_DIM + o];
#pragma unroll 8
    for (int k = 0; k < H_DIM; ++k) acc += fmaxf(h2[k], 0.f) * Wl[(H_DIM + k) * OUT_DIM + o];
#pragma unroll
    for (int k = 0; k < F_IN; ++k)  acc += fmaxf(xr[k], 0.f) * Wl[(2 * H_DIM + k) * OUT_DIM + o];
    out[idx] = acc;
}

// --------------------------------------------------------------- launcher
extern "C" void kernel_launch(void* const* d_in, const int* in_sizes, int n_in,
                              void* d_out, int out_size, void* d_ws, size_t ws_size,
                              hipStream_t stream) {
    const float* x     = (const float*)d_in[0];
    const int*   ei    = (const int*)d_in[1];
    const float* ew    = (const float*)d_in[2];
    const float* W1    = (const float*)d_in[3];
    const float* b1    = (const float*)d_in[4];
    const float* W2    = (const float*)d_in[5];
    const float* b2    = (const float*)d_in[6];
    const float* g1    = (const float*)d_in[7];
    const float* be1   = (const float*)d_in[8];
    const float* g2    = (const float*)d_in[9];
    const float* be2   = (const float*)d_in[10];
    const float* W_ih1 = (const float*)d_in[11];
    // d_in[12] = W_hh1 (unused: h0 = 0)
    const float* b_ih1 = (const float*)d_in[13];
    const float* b_hh1 = (const float*)d_in[14];
    const float* W_ih2 = (const float*)d_in[15];
    // d_in[16] = W_hh2 (unused: h0 = 0)
    const float* b_ih2 = (const float*)d_in[17];
    const float* b_hh2 = (const float*)d_in[18];
    const float* Wl    = (const float*)d_in[19];
    const float* bl    = (const float*)d_in[20];
    float* out = (float*)d_out;

    // workspace partition (floats)
    float* ws    = (float*)d_ws;
    float* deg   = ws;                    // N
    float* dinv  = deg + N_NODES;         // N
    float* dinv2 = dinv + N_NODES;        // N
    float* xw    = dinv2 + N_NODES;       // N*H
    float* agg   = xw + N_NODES * H_DIM;  // N*H (becomes h_pre)
    float* rbuf  = agg + N_NODES * H_DIM; // N*2H (concat h1|h2)
    float* hh1   = rbuf + 2 * N_NODES * H_DIM;  // N*H
    float* hh2   = hh1 + N_NODES * H_DIM;       // N*H
    float* cs1   = hh2 + N_NODES * H_DIM;       // 64
    float* cq1   = cs1 + H_DIM;                 // 64
    float* cs2   = cq1 + H_DIM;                 // 64
    float* cq2   = cs2 + H_DIM;                 // 64
    // packed f16 weights (32B-aligned: all offsets below are multiples of 8 floats)
    _Float16* BpW1 = (_Float16*)(cq2 + H_DIM);      // 4 tiles * 1 chunk  = 2048 halfs
    _Float16* BpW2 = BpW1 + 4 * 1 * 32 * 16;        // 4 tiles * 2 chunks = 4096 halfs
    _Float16* BpL1 = BpW2 + 4 * 2 * 32 * 16;        // 16 tiles * 4 chunks = 32768 halfs
    _Float16* BpL2 = BpL1 + 16 * 4 * 32 * 16;       // 16 tiles * 2 chunks = 16384 halfs

    const int TPB = 256;
    const int NH  = N_NODES * H_DIM;                         // 6,400,000
    const int gemmBlocks = ((N_NODES / 16) + 7) / 8;         // 1 wave = 16x64 strip
    const int lstmBlocks = (N_NODES / 16) * (H_DIM / 16) / 8;  // 3125

    // weight packing (tiny, L2-resident)
    k_pack_b<<<(4 * 1 * 512 + TPB - 1) / TPB, TPB, 0, stream>>>(W1, BpW1, F_IN, H_DIM, H_DIM, 0);
    k_pack_b<<<(4 * 2 * 512 + TPB - 1) / TPB, TPB, 0, stream>>>(W2, BpW2, H_DIM, H_DIM, H_DIM, 0);
    k_pack_b<<<(16 * 4 * 512 + TPB - 1) / TPB, TPB, 0, stream>>>(W_ih1, BpL1, 2 * H_DIM, 2 * H_DIM, 4 * H_DIM, 1);
    k_pack_b<<<(16 * 2 * 512 + TPB - 1) / TPB, TPB, 0, stream>>>(W_ih2, BpL2, H_DIM, H_DIM, 4 * H_DIM, 1);

    // degree + norm
    k_zero<<<(N_NODES + TPB - 1) / TPB, TPB, 0, stream>>>(deg, N_NODES);
    k_zero<<<1, TPB, 0, stream>>>(cs1, 4 * H_DIM);  // cs1|cq1|cs2|cq2 contiguous
    k_zero<<<(NH + TPB - 1) / TPB, TPB, 0, stream>>>(agg, NH);
    k_deg<<<(N_EDGES + TPB - 1) / TPB, TPB, 0, stream>>>(ei, ew, deg);
    k_dinv<<<(N_NODES + TPB - 1) / TPB, TPB, 0, stream>>>(deg, dinv, dinv2);

    // ---- GCN layer 1 (K = 15: guarded scalar A path)
    k_gemm_wmma<true><<<gemmBlocks, TPB, 0, stream>>>(x, BpW1, xw, F_IN, F_IN, H_DIM);
    k_edge_scatter<<<(N_EDGES * 16) / TPB, TPB, 0, stream>>>(ei, ew, dinv, xw, agg);
    k_post_gcn<<<NH / TPB, TPB, 0, stream>>>(agg, xw, dinv2, b1, cs1, cq1);
    k_bn_apply<<<NH / TPB, TPB, 0, stream>>>(agg, cs1, cq1, g1, be1, rbuf, 2 * H_DIM, 0);

    // ---- GCN layer 2 (K = 64, lda = 128: vector A path)
    k_zero<<<(NH + TPB - 1) / TPB, TPB, 0, stream>>>(agg, NH);
    k_gemm_wmma<false><<<gemmBlocks, TPB, 0, stream>>>(rbuf, BpW2, xw, H_DIM, 2 * H_DIM, H_DIM);
    k_edge_scatter<<<(N_EDGES * 16) / TPB, TPB, 0, stream>>>(ei, ew, dinv, xw, agg);
    k_post_gcn<<<NH / TPB, TPB, 0, stream>>>(agg, xw, dinv2, b2, cs2, cq2);
    k_bn_apply<<<NH / TPB, TPB, 0, stream>>>(agg, cs2, cq2, g2, be2, rbuf, 2 * H_DIM, H_DIM);

    // ---- LSTM steps (async-LDS staged weights + fused WMMA gates)
    k_lstm_wmma<<<lstmBlocks, TPB, 0, stream>>>(rbuf, 2 * H_DIM, 2 * H_DIM, BpL1, b_ih1, b_hh1, hh1);
    k_lstm_wmma<<<lstmBlocks, TPB, 0, stream>>>(hh1, H_DIM, H_DIM, BpL2, b_ih2, b_hh2, hh2);

    // ---- output projection
    k_final<<<(N_NODES * OUT_DIM + TPB - 1) / TPB, TPB, 0, stream>>>(
        hh1, hh2, x, Wl, bl, out);
}
